// NODE_TEXTCNN_2980707303908
// MI455X (gfx1250) — compile-verified, matrix operand order checked by reading, not run on hardware
//
#include <hip/hip_runtime.h>
#include <hip/hip_bf16.h>

// ---------------------------------------------------------------------------
// NODE-TextCNN for MI455X (gfx1250, wave32, WMMA bf16->f32, async-to-LDS)
//
//   0) pack w_ode / conv weights into WMMA B-fragment layout (bf16,
//      split-half planes -> 16B lane stride for ds/global b128 loads)
//   1) fused gather + RK4 ODE: 64-token tile resident in LDS, 40 f-evals
//      via v_wmma_f32_16x16x32_bf16.  w_ode K-chunks double-buffered into
//      LDS with global_load_async_to_lds_b128 (ASYNCcnt) overlapping the
//      128 WMMAs of the previous chunk.  B fragments are software-pipelined
//      two-deep in registers so ds_load latency hides under the WMMAs.
//   2) convs as shifted GEMMs via WMMA -> c_pre (f32)
//   3) per-channel batch stats, 4) BN+ReLU+maxpool, 5) classifier
// ---------------------------------------------------------------------------

typedef __attribute__((ext_vector_type(16))) __bf16 v16bf;
typedef __attribute__((ext_vector_type(8)))  __bf16 v8bf;
typedef __attribute__((ext_vector_type(8)))  float  v8f;

#define D_DIM  512
#define N_TOK  32768          // 32 * 1024
#define MT     64             // tokens per ODE workgroup tile
#define NSTEP  10             // ODE_STEP - 1
#define C_CONV 128
#define B_SZ   32
#define S_LEN  1024

// ---------------------------------------------------------------------------
// B-fragment layout for v_wmma_*_16x16x32_bf16 (512 bf16 = 1KB per fragment).
// Fragment (kb, ntg) holds B(k,n), k in [kb*32,+32), n in [ntg*16,+16).
//   pos = h*256 + lane*8 + e ; K = kb*32 + (lane>>4)*8 + h*16 + e ;
//   n = ntg*16 + (lane&15).
// ---------------------------------------------------------------------------
__global__ void pack_wode_kernel(const float* __restrict__ w, __bf16* __restrict__ wpk) {
    int id = blockIdx.x * blockDim.x + threadIdx.x;      // 512*512 elements
    if (id >= D_DIM * D_DIM) return;
    int pos  = id & 511;
    int frag = id >> 9;
    int e    = pos & 7;
    int lane = (pos >> 3) & 31;
    int h    = pos >> 8;
    int kb   = frag >> 5;          // 16 K-blocks
    int ntg  = frag & 31;          // 32 N-tiles
    int k = kb * 32 + ((lane >> 4) * 8) + h * 16 + e;
    int n = ntg * 16 + (lane & 15);
    wpk[id] = (__bf16)w[n * D_DIM + k];        // B(k,n) = w[n][k]
}

__global__ void pack_cw_kernel(const float* __restrict__ cw, __bf16* __restrict__ pk, int KW) {
    int id = blockIdx.x * blockDim.x + threadIdx.x;      // KW*16*8*512 elements
    if (id >= KW * 16 * 8 * 512) return;
    int pos  = id & 511;
    int frag = id >> 9;
    int e    = pos & 7;
    int lane = (pos >> 3) & 31;
    int h    = pos >> 8;
    int ntg  = frag & 7;
    int kb   = (frag >> 3) & 15;
    int j    = frag >> 7;                                // conv tap
    int d  = kb * 32 + ((lane >> 4) * 8) + h * 16 + e;
    int oc = ntg * 16 + (lane & 15);
    pk[id] = (__bf16)cw[(oc * D_DIM + d) * KW + j];      // B_j(d,oc) = cw[oc][d][j]
}

__global__ void zero_stats_kernel(float* __restrict__ stats) {
    int i = blockIdx.x * blockDim.x + threadIdx.x;
    if (i < 3 * C_CONV * 2) stats[i] = 0.0f;
}

// ---------------------------------------------------------------------------
// Async stage of one 32KB w_ode K-chunk (32 fragments) into LDS.
// 256 threads x 128B, 8 x b128 async copies per thread (ASYNCcnt <= 8).
// ---------------------------------------------------------------------------
__device__ __forceinline__ void stage_b(const __bf16* __restrict__ wpk, int kb,
                                        __bf16* dst, int tid)
{
    unsigned lds0 = (unsigned)(unsigned long long)dst + (unsigned)tid * 128u;
    unsigned long long g0 =
        (unsigned long long)(wpk + (size_t)kb * 32 * 512) + (unsigned long long)tid * 128ull;
    #pragma unroll
    for (int i = 0; i < 8; ++i) {
        asm volatile("global_load_async_to_lds_b128 %0, %1, off offset:%2"
                     :: "v"(lds0), "v"(g0), "i"(i * 16) : "memory");
    }
}

__device__ __forceinline__ void wait_async0() {
    asm volatile("s_wait_asynccnt 0x0" ::: "memory");
}

// ---------------------------------------------------------------------------
// Fused embedding gather + RK4 ODE.  One workgroup = 64 tokens, 8 waves.
// Wave (mb, nh): mb = M-block (4 x 16 rows), nh = N-half (2 x 256 cols).
// hn / acc live in WMMA C/D layout: VGPR j <-> row mb*16 + j + 8*(lane>>4),
// column nh*256 + nt*16 + (lane&15).
// ---------------------------------------------------------------------------
__global__ __launch_bounds__(256, 1) void ode_rk4_kernel(
    const int* __restrict__ ids, const float* __restrict__ emb,
    const __bf16* __restrict__ wpk, const float* __restrict__ bvec,
    __bf16* __restrict__ hout)
{
    __shared__ float  h_s[MT][D_DIM + 4];     // f32 state
    __shared__ __bf16 z_s[MT][D_DIM + 8];     // bf16 WMMA A source
    __shared__ __bf16 bbuf[2][32 * 512];      // double-buffered w_ode K-chunks

    const int tid  = threadIdx.x;
    const int lane = tid & 31;
    const int wave = tid >> 5;
    const int mb   = wave & 3;
    const int nh   = wave >> 2;
    const int wg   = blockIdx.x;

    // ---- embedding gather: 4 threads per row, float4 ----
    {
        const int r  = tid >> 2;
        const int c0 = (tid & 3) * 128;
        const int id = ids[wg * MT + r];
        const float4* src = (const float4*)(emb + (size_t)id * D_DIM + c0);
        #pragma unroll 4
        for (int i = 0; i < 32; ++i) {
            float4 v = src[i];
            int c = c0 + i * 4;
            h_s[r][c+0] = v.x; h_s[r][c+1] = v.y; h_s[r][c+2] = v.z; h_s[r][c+3] = v.w;
            z_s[r][c+0] = (__bf16)v.x; z_s[r][c+1] = (__bf16)v.y;
            z_s[r][c+2] = (__bf16)v.z; z_s[r][c+3] = (__bf16)v.w;
        }
    }
    __syncthreads();

    const int mrow = mb * 16 + ((lane >> 4) * 8);
    const int ncol = nh * 256 + (lane & 15);

    v8f   hn[16];
    float bias[16];
    #pragma unroll
    for (int nt = 0; nt < 16; ++nt) {
        bias[nt] = bvec[ncol + nt * 16];
        #pragma unroll
        for (int j = 0; j < 8; ++j) hn[nt][j] = h_s[mrow + j][ncol + nt * 16];
    }

    const float dt = 0.1f;
    for (int step = 0; step < NSTEP; ++step) {
        for (int ev = 0; ev < 4; ++ev) {
            v8f acc[16];
            #pragma unroll
            for (int nt = 0; nt < 16; ++nt)
                #pragma unroll
                for (int j = 0; j < 8; ++j) acc[nt][j] = bias[nt];

            // prime K-chunk 0
            stage_b(wpk, 0, &bbuf[0][0], tid);
            wait_async0();
            __syncthreads();

            const int arow = mb * 16 + (lane & 15);
            for (int kb = 0; kb < 16; ++kb) {
                if (kb < 15) stage_b(wpk, kb + 1, &bbuf[(kb + 1) & 1][0], tid);

                // A fragment (16x32 bf16) from row-major z_s
                const int kbase = kb * 32 + ((lane >> 4) * 8);
                v8bf lo = *(const v8bf*)&z_s[arow][kbase];
                v8bf hi = *(const v8bf*)&z_s[arow][kbase + 16];
                v16bf a;
                #pragma unroll
                for (int e = 0; e < 8; ++e) { a[e] = lo[e]; a[e + 8] = hi[e]; }

                const __bf16* bb = &bbuf[kb & 1][0] + ((size_t)(nh * 16) << 9) + lane * 8;
                auto loadB = [&](int nt) {
                    const __bf16* fb = bb + ((size_t)nt << 9);
                    v8bf bl = *(const v8bf*)fb;
                    v8bf bh = *(const v8bf*)(fb + 256);
                    v16bf bf;
                    #pragma unroll
                    for (int e = 0; e < 8; ++e) { bf[e] = bl[e]; bf[e + 8] = bh[e]; }
                    return bf;
                };

                // two-deep register pipeline: load(nt+1) issues before wmma(nt)
                v16bf bcur = loadB(0);
                #pragma unroll
                for (int nt = 0; nt < 16; ++nt) {
                    v16bf bnxt;
                    if (nt < 15) bnxt = loadB(nt + 1);
                    acc[nt] = __builtin_amdgcn_wmma_f32_16x16x32_bf16(
                        false, a, false, bcur, (short)0, acc[nt], false, false);
                    if (nt < 15) bcur = bnxt;
                }

                wait_async0();       // next chunk landed (overlapped with WMMAs)
                __syncthreads();     // also fences z_s reads on last chunk
            }

            const float wB = (ev == 0 || ev == 3) ? dt / 6.0f : dt / 3.0f;
            if (ev < 3) {
                const float wC = (ev < 2) ? 0.5f * dt : dt;
                #pragma unroll
                for (int nt = 0; nt < 16; ++nt)
                    #pragma unroll
                    for (int j = 0; j < 8; ++j) {
                        float k = acc[nt][j];
                        hn[nt][j] += wB * k;
                        int rr = mrow + j, cc = ncol + nt * 16;
                        z_s[rr][cc] = (__bf16)(h_s[rr][cc] + wC * k);
                    }
            } else {
                #pragma unroll
                for (int nt = 0; nt < 16; ++nt)
                    #pragma unroll
                    for (int j = 0; j < 8; ++j) {
                        float v = hn[nt][j] + wB * acc[nt][j];
                        hn[nt][j] = v;
                        int rr = mrow + j, cc = ncol + nt * 16;
                        h_s[rr][cc] = v;
                        z_s[rr][cc] = (__bf16)v;
                    }
            }
            __syncthreads();
        }
    }

    // ---- write final h, bf16 row-major [32768][512] ----
    #pragma unroll
    for (int nt = 0; nt < 16; ++nt)
        #pragma unroll
        for (int j = 0; j < 8; ++j)
            hout[((size_t)(wg * MT + mrow + j) << 9) + (ncol + nt * 16)] = (__bf16)hn[nt][j];
}

// ---------------------------------------------------------------------------
// Conv as KW shifted GEMMs: rows r=(b*1024+s), A = h rows r+j (bf16),
// B = packed conv weights (global, L2-resident).  One wave = 16 rows x 128 ch.
// Rows with s >= S_out compute garbage and are excluded later.
// ---------------------------------------------------------------------------
template <int KW>
__global__ __launch_bounds__(256, 1) void conv_gemm_kernel(
    const __bf16* __restrict__ h, const __bf16* __restrict__ cwpk,
    const float* __restrict__ cb, float* __restrict__ cpre)
{
    const int tid  = threadIdx.x;
    const int lane = tid & 31;
    const int wave = tid >> 5;
    const long rbase = ((long)blockIdx.x * 8 + wave) * 16;

    v8f   acc[8];
    float bias[8];
    #pragma unroll
    for (int nt = 0; nt < 8; ++nt) {
        bias[nt] = cb[nt * 16 + (lane & 15)];
        #pragma unroll
        for (int j = 0; j < 8; ++j) acc[nt][j] = bias[nt];
    }

    #pragma unroll
    for (int j = 0; j < KW; ++j) {
        for (int kb = 0; kb < 16; ++kb) {
            long arow = rbase + (lane & 15) + j;
            if (arow > (long)N_TOK - 1) arow = N_TOK - 1;   // clamp; masked later
            const int kbase = kb * 32 + ((lane >> 4) * 8);
            const __bf16* ap = h + (arow << 9) + kbase;
            v8bf lo = *(const v8bf*)ap;
            v8bf hi = *(const v8bf*)(ap + 16);
            v16bf a;
            #pragma unroll
            for (int e = 0; e < 8; ++e) { a[e] = lo[e]; a[e + 8] = hi[e]; }

            const __bf16* bbase = cwpk + ((size_t)((j * 16 + kb) * 8) << 9) + lane * 8;
            auto loadB = [&](int nt) {
                const __bf16* fb = bbase + ((size_t)nt << 9);
                v8bf bl = *(const v8bf*)fb;
                v8bf bh = *(const v8bf*)(fb + 256);
                v16bf bf;
                #pragma unroll
                for (int e = 0; e < 8; ++e) { bf[e] = bl[e]; bf[e + 8] = bh[e]; }
                return bf;
            };

            v16bf bcur = loadB(0);
            #pragma unroll
            for (int nt = 0; nt < 8; ++nt) {
                v16bf bnxt;
                if (nt < 7) bnxt = loadB(nt + 1);
                acc[nt] = __builtin_amdgcn_wmma_f32_16x16x32_bf16(
                    false, a, false, bcur, (short)0, acc[nt], false, false);
                if (nt < 7) bcur = bnxt;
            }
        }
    }

    const int mrow = (lane >> 4) * 8;
    #pragma unroll
    for (int nt = 0; nt < 8; ++nt)
        #pragma unroll
        for (int j = 0; j < 8; ++j)
            cpre[((size_t)(rbase + mrow + j) << 7) + nt * 16 + (lane & 15)] = acc[nt][j];
}

// ---------------------------------------------------------------------------
__global__ void stats_kernel(const float* __restrict__ cpre, float* __restrict__ stats, int sout)
{
    const int c = blockIdx.x, b = blockIdx.y, tid = threadIdx.x;
    float s = 0.f, s2 = 0.f;
    for (int i = tid; i < sout; i += 256) {
        float v = cpre[(((size_t)b * S_LEN + i) << 7) + c];
        s += v; s2 += v * v;
    }
    __shared__ float rs[256], rq[256];
    rs[tid] = s; rq[tid] = s2;
    __syncthreads();
    for (int off = 128; off > 0; off >>= 1) {
        if (tid < off) { rs[tid] += rs[tid + off]; rq[tid] += rq[tid + off]; }
        __syncthreads();
    }
    if (tid == 0) {
        atomicAdd(&stats[c * 2 + 0], rs[0]);
        atomicAdd(&stats[c * 2 + 1], rq[0]);
    }
}

__global__ void pool_kernel(const float* __restrict__ cpre, const float* __restrict__ stats,
                            const float* __restrict__ g, const float* __restrict__ bt,
                            float* __restrict__ feats, int sout, int featoff)
{
    const int c = blockIdx.x, b = blockIdx.y, tid = threadIdx.x;
    float mx = -1e30f, mn = 1e30f;
    for (int i = tid; i < sout; i += 256) {
        float v = cpre[(((size_t)b * S_LEN + i) << 7) + c];
        mx = fmaxf(mx, v); mn = fminf(mn, v);
    }
    __shared__ float rmx[256], rmn[256];
    rmx[tid] = mx; rmn[tid] = mn;
    __syncthreads();
    for (int off = 128; off > 0; off >>= 1) {
        if (tid < off) {
            rmx[tid] = fmaxf(rmx[tid], rmx[tid + off]);
            rmn[tid] = fminf(rmn[tid], rmn[tid + off]);
        }
        __syncthreads();
    }
    if (tid == 0) {
        float n   = (float)B_SZ * (float)sout;
        float mu  = stats[c * 2 + 0] / n;
        float var = stats[c * 2 + 1] / n - mu * mu;
        float sc  = g[c] * rsqrtf(var + 1e-5f);
        float sel = (sc >= 0.f) ? rmx[0] : rmn[0];
        float val = sc * (sel - mu) + bt[c];
        feats[b * 384 + featoff + c] = fmaxf(val, 0.f);
    }
}

__global__ void cls_kernel(const float* __restrict__ feats, const float* __restrict__ w,
                           const float* __restrict__ bias, float* __restrict__ out)
{
    int i = blockIdx.x * blockDim.x + threadIdx.x;
    if (i >= B_SZ * 10) return;
    int b = i / 10, n = i % 10;
    float s = bias[n];
    for (int d = 0; d < 384; ++d) s += feats[b * 384 + d] * w[n * 384 + d];
    out[b * 10 + n] = s;
}

// ---------------------------------------------------------------------------
extern "C" void kernel_launch(void* const* d_in, const int* in_sizes, int n_in,
                              void* d_out, int out_size, void* d_ws, size_t ws_size,
                              hipStream_t stream)
{
    const int*   ids   = (const int*)  d_in[0];
    const float* emb   = (const float*)d_in[1];
    const float* w_ode = (const float*)d_in[2];
    const float* b_ode = (const float*)d_in[3];
    const float* cw[3] = {(const float*)d_in[4],  (const float*)d_in[8],  (const float*)d_in[12]};
    const float* cb[3] = {(const float*)d_in[5],  (const float*)d_in[9],  (const float*)d_in[13]};
    const float* g [3] = {(const float*)d_in[6],  (const float*)d_in[10], (const float*)d_in[14]};
    const float* bt[3] = {(const float*)d_in[7],  (const float*)d_in[11], (const float*)d_in[15]};
    const float* wcls  = (const float*)d_in[16];
    const float* bcls  = (const float*)d_in[17];
    float* out = (float*)d_out;

    const int KW[3]   = {3, 4, 5};
    const int SOUT[3] = {S_LEN - 3 + 1, S_LEN - 4 + 1, S_LEN - 5 + 1};

    // ---- workspace carve-up ----
    char* ws = (char*)d_ws;
    size_t off = 0;
    auto carve = [&](size_t bytes) { size_t o = off; off += (bytes + 255) & ~(size_t)255; return o; };
    __bf16* h_bf = (__bf16*)(ws + carve((size_t)N_TOK * D_DIM * 2));     // 32 MB
    __bf16* wpk  = (__bf16*)(ws + carve((size_t)D_DIM * D_DIM * 2));     // 0.5 MB
    __bf16* cwpk[3];
    for (int i = 0; i < 3; ++i)
        cwpk[i] = (__bf16*)(ws + carve((size_t)KW[i] * 16 * 8 * 512 * 2));
    float* cpre[3];
    for (int i = 0; i < 3; ++i)
        cpre[i] = (float*)(ws + carve((size_t)N_TOK * C_CONV * 4));      // 16 MB each
    float* stats = (float*)(ws + carve(3 * C_CONV * 2 * sizeof(float)));
    float* feats = (float*)(ws + carve((size_t)B_SZ * 384 * sizeof(float)));

    // ---- 0: zero stats + pack weights ----
    zero_stats_kernel<<<3, 256, 0, stream>>>(stats);
    pack_wode_kernel<<<(D_DIM * D_DIM + 255) / 256, 256, 0, stream>>>(w_ode, wpk);
    for (int i = 0; i < 3; ++i) {
        int elems = KW[i] * 16 * 8 * 512;
        pack_cw_kernel<<<(elems + 255) / 256, 256, 0, stream>>>(cw[i], cwpk[i], KW[i]);
    }

    // ---- 1: fused gather + RK4 ODE (the 687 GFLOP hot loop) ----
    ode_rk4_kernel<<<N_TOK / MT, 256, 0, stream>>>(ids, emb, wpk, b_ode, h_bf);

    // ---- 2: convs ----
    conv_gemm_kernel<3><<<N_TOK / 128, 256, 0, stream>>>(h_bf, cwpk[0], cb[0], cpre[0]);
    conv_gemm_kernel<4><<<N_TOK / 128, 256, 0, stream>>>(h_bf, cwpk[1], cb[1], cpre[1]);
    conv_gemm_kernel<5><<<N_TOK / 128, 256, 0, stream>>>(h_bf, cwpk[2], cb[2], cpre[2]);

    // ---- 3: batch stats ----
    for (int i = 0; i < 3; ++i)
        stats_kernel<<<dim3(C_CONV, B_SZ), 256, 0, stream>>>(cpre[i], stats + i * C_CONV * 2, SOUT[i]);

    // ---- 4: BN + ReLU + max pool ----
    for (int i = 0; i < 3; ++i)
        pool_kernel<<<dim3(C_CONV, B_SZ), 256, 0, stream>>>(
            cpre[i], stats + i * C_CONV * 2, g[i], bt[i], feats, SOUT[i], i * C_CONV);

    // ---- 5: classifier ----
    cls_kernel<<<(B_SZ * 10 + 255) / 256, 256, 0, stream>>>(feats, wcls, bcls, out);
}
